// Pytorch_DTLN_stateful_11501922418992
// MI455X (gfx1250) — matrix operand order repaired and checked by reading, not compiled, not run on hardware
//
#include <hip/hip_runtime.h>
#include <hip/hip_bf16.h>

// ---------------------------------------------------------------------------
// DTLN on MI455X (gfx1250): everything heavy is V_WMMA_F32_16X16X32_F16.
//  - rfft/irfft realized as f16 WMMA GEMMs vs on-device DFT matrices
//  - GEMM: wave = 16x64 tile, B-panel staged in LDS (double buffered),
//    A streamed from global (L2-resident: A operand <= 135MB < 192MB L2)
//  - LSTM recurrence: per-WG persistent scan; Whh held in REGISTERS
//    (16 loop-invariant WMMA B-fragments), state + gate buffer in LDS
//  - gate nonlinearities use single-instruction trans ops (v_exp/v_rcp/
//    v_tanh) instead of the precise-division sigmoid expansion
// ---------------------------------------------------------------------------

typedef __attribute__((ext_vector_type(16))) _Float16 v16h;
typedef __attribute__((ext_vector_type(8)))  float    v8f;

union V8F  { float f[8]; v8f v; };

#define TWO_PI_OVER_1024 6.1359231515425649e-3f

// --- fragment loaders (contiguous k order, same for A and B). Work for global or LDS ptrs. ---
// A 16x32 f16: lane -> row=lane%16; elems 0..15 = K[laneHi*16 + 0..15]
//              (two b128 loads)
__device__ __forceinline__ v16h load_frag_a(const _Float16* p) {
  union { uint4 q[2]; v16h h; } u;
  u.q[0] = *reinterpret_cast<const uint4*>(p);
  u.q[1] = *reinterpret_cast<const uint4*>(p + 8);
  return u.h;
}
// B 32x16 f16 stored as Bt (N x K row-major): lane -> col n=lane%16;
// elems 0..15 = K[laneHi*16 + 0..15] (two b128 loads)
__device__ __forceinline__ v16h load_frag_b(const _Float16* p) {
  union { uint4 q[2]; v16h h; } u;
  u.q[0] = *reinterpret_cast<const uint4*>(p);
  u.q[1] = *reinterpret_cast<const uint4*>(p + 8);
  return u.h;
}

// fast sigmoid: v_exp_f32 + v_rcp_f32 (avoids precise-div expansion)
__device__ __forceinline__ float sigm(float x) {
  return __builtin_amdgcn_rcpf(1.f + __expf(-x));
}
// fast tanh: gfx1250 has V_TANH_F32; fall back to exp+rcp algebra
__device__ __forceinline__ float tanh_fast(float x) {
#if __has_builtin(__builtin_amdgcn_tanhf)
  return __builtin_amdgcn_tanhf(x);
#else
  return 2.f * __builtin_amdgcn_rcpf(1.f + __expf(-2.f * x)) - 1.f;
#endif
}

// ---------------------------------------------------------------------------
// Generic WMMA GEMM: C(MxN) = act(A(MxK) * Bt(NxK)^T + bias)
// block = 256 thr (8 waves); wave tile 16(M) x 64(N); K stepped by 32.
// B-panel (64x32 f16 = 4KB) cooperatively staged in LDS, double buffered:
// one barrier per K-step; all 8 waves share the staged panel (8x less vmem).
// ACT: 0 none, 1 sigmoid. OUT16: 1 -> f16 output, 0 -> f32.
// Requires: M%16==0, K%32==0, N%16==0, lda/ldb multiples of 8 halves.
// ---------------------------------------------------------------------------
template <int ACT, int OUT16>
__global__ __launch_bounds__(256) void gemm_wmma(
    const _Float16* __restrict__ A, int lda,
    const _Float16* __restrict__ Bt, int ldb,
    const float* __restrict__ bias,
    void* __restrict__ Cout, int ldc,
    int M, int N, int K)
{
  __shared__ _Float16 sB[2][64 * 32];   // double-buffered B panel (8KB)

  const int tid    = threadIdx.x;
  const int lane   = tid & 31;
  const int wv     = tid >> 5;
  const int laneLo = lane & 15;
  const int laneHi = lane >> 4;
  const int m0     = blockIdx.y * 128 + wv * 16;
  const bool mvalid = (m0 < M);
  const int mA     = mvalid ? m0 : 0;       // invalid waves compute on row 0
  const int nBase  = blockIdx.x * 64;

  // staging map: thread -> (row 0..63, 8-half chunk 0..3) of the 64x32 panel
  const int sr = tid >> 2;
  const int sc = (tid & 3) * 8;
  const int gn = nBase + sr;

  V8F acc[4];
#pragma unroll
  for (int j = 0; j < 4; ++j)
#pragma unroll
    for (int r = 0; r < 8; ++r) acc[j].f[r] = 0.f;

  const _Float16* aRow = A + (size_t)(mA + laneLo) * lda + laneHi * 16;

  // prologue: stage panel for k=0 (zero-fill rows beyond N)
  {
    uint4 v = make_uint4(0u, 0u, 0u, 0u);
    if (gn < N) v = *reinterpret_cast<const uint4*>(Bt + (size_t)gn * ldb + sc);
    *reinterpret_cast<uint4*>(&sB[0][sr * 32 + sc]) = v;
  }
  __syncthreads();

  for (int k = 0; k < K; k += 32) {
    const int cur = (k >> 5) & 1;
    if (k + 32 < K) {                       // stage next panel into other buf
      uint4 v = make_uint4(0u, 0u, 0u, 0u);
      if (gn < N)
        v = *reinterpret_cast<const uint4*>(Bt + (size_t)gn * ldb + (k + 32) + sc);
      *reinterpret_cast<uint4*>(&sB[cur ^ 1][sr * 32 + sc]) = v;
    }
    v16h a = load_frag_a(aRow + k);
    __builtin_prefetch(aRow + k + 64, 0, 1);   // global_prefetch for A stream
#pragma unroll
    for (int j = 0; j < 4; ++j) {
      v16h b = load_frag_b(&sB[cur][(j * 16 + laneLo) * 32 + laneHi * 16]);
      acc[j].v = __builtin_amdgcn_wmma_f32_16x16x32_f16(
          false, a, false, b, (short)0, acc[j].v, false, false);
    }
    __syncthreads();
  }

#pragma unroll
  for (int j = 0; j < 4; ++j) {
    const int n0 = nBase + j * 16;
    if (n0 >= N) break;
    if (!mvalid) continue;
    const int col = n0 + laneLo;
    const float bv = bias ? bias[col] : 0.f;
#pragma unroll
    for (int r = 0; r < 8; ++r) {
      const int row = m0 + laneHi * 8 + r;
      float v = acc[j].f[r] + bv;
      if (ACT == 1) v = sigm(v);
      if (OUT16) ((_Float16*)Cout)[(size_t)row * ldc + col] = (_Float16)v;
      else       ((float*)Cout)[(size_t)row * ldc + col] = v;
    }
  }
}

// ---------------------------------------------------------------------------
// LSTM scan: grid = B/16 blocks, 256 thr (8 waves). Each block owns 16 batch
// rows for all T=997 steps. Whh is loop-invariant -> each wave preloads its
// 16 B-fragments (64 gate cols x K=128) into 128 VGPRs once; per step only
// the h A-fragments come from LDS (ds_load_b128) and 16 WMMAs issue with 4
// independent accumulator chains (hides the f16 WMMA RAW hazard window).
// LDS: gate buf (16x512 f32) + h (16x128 f16) + c (16x128 f32) = 44KB.
// ---------------------------------------------------------------------------
__global__ __launch_bounds__(256) void lstm_scan(
    const float* __restrict__ xg,       // (B*T, 512), row = b*T + t
    const _Float16* __restrict__ Whh,   // (512,128) row-major [gate][hid]
    _Float16* __restrict__ hsOut,       // (B*T, 128)
    int T)
{
  __shared__ float    sG[16 * 512];
  __shared__ _Float16 sH[16 * 128];
  __shared__ float    sC[16 * 128];

  const int tid    = threadIdx.x;
  const int lane   = tid & 31;
  const int wv     = tid >> 5;
  const int laneLo = lane & 15;
  const int laneHi = lane >> 4;
  const int bb     = blockIdx.x * 16;

  // preload this wave's 16 loop-invariant Whh fragments into registers
  v16h bfr[4][4];
#pragma unroll
  for (int kk4 = 0; kk4 < 4; ++kk4)
#pragma unroll
    for (int j = 0; j < 4; ++j)
      bfr[kk4][j] = load_frag_b(Whh + (size_t)(wv * 64 + j * 16 + laneLo) * 128
                                + kk4 * 32 + laneHi * 16);

  for (int i = tid; i < 16 * 128; i += 256) { sH[i] = (_Float16)0.f; sC[i] = 0.f; }
  __syncthreads();

  for (int t = 0; t < T; ++t) {
    // ---- g-tile: wave wv computes cols [wv*64, wv*64+64) of (16 x 512) ----
    V8F acc[4];
#pragma unroll
    for (int j = 0; j < 4; ++j)
#pragma unroll
      for (int r = 0; r < 8; ++r)
        acc[j].f[r] = xg[((size_t)(bb + laneHi * 8 + r) * T + t) * 512
                         + wv * 64 + j * 16 + laneLo];

#pragma unroll
    for (int kk4 = 0; kk4 < 4; ++kk4) {
      v16h a = load_frag_a(sH + laneLo * 128 + kk4 * 32 + laneHi * 16);
#pragma unroll
      for (int j = 0; j < 4; ++j)
        acc[j].v = __builtin_amdgcn_wmma_f32_16x16x32_f16(
            false, a, false, bfr[kk4][j], (short)0, acc[j].v, false, false);
    }
#pragma unroll
    for (int j = 0; j < 4; ++j)
#pragma unroll
      for (int r = 0; r < 8; ++r)
        sG[(laneHi * 8 + r) * 512 + wv * 64 + j * 16 + laneLo] = acc[j].f[r];
    __syncthreads();

    // ---- gates: i,f,g,o chunks of 128; update c,h; stream h out ----
    for (int i = tid; i < 16 * 128; i += 256) {
      const int m = i >> 7, jj = i & 127;
      const float gi = sG[m * 512 + jj];
      const float gf = sG[m * 512 + 128 + jj];
      const float gg = sG[m * 512 + 256 + jj];
      const float go = sG[m * 512 + 384 + jj];
      float c = sigm(gf) * sC[i] + sigm(gi) * tanh_fast(gg);
      float h = sigm(go) * tanh_fast(c);
      sC[i] = c;
      sH[i] = (_Float16)h;
      hsOut[((size_t)(bb + m) * T + t) * 128 + jj] = (_Float16)h;
    }
    __syncthreads();
  }
}

// ---------------------------------------------------------------------------
// Elementwise / setup kernels
// ---------------------------------------------------------------------------
__global__ void k_frames(const float* __restrict__ x, _Float16* __restrict__ fr,
                         int T, int NS, size_t total)
{
  size_t i = (size_t)blockIdx.x * 256 + threadIdx.x;
  if (i >= total) return;
  const int j = (int)(i & 1023);
  const size_t r = i >> 10;
  const int t = (int)(r % T);
  const int b = (int)(r / T);
  const float w = 0.5f * (1.0f - __cosf(6.28318530717958647f * (float)j * (1.f / 1024.f)));
  fr[i] = (_Float16)(x[(size_t)b * NS + (size_t)t * 256 + j] * w);
}

// spec layout: ld 1056; cols [0,513) real, [513,1026) imag, rest zero-padded
__global__ void k_mag(const _Float16* __restrict__ spec, _Float16* __restrict__ mag,
                      size_t total)
{
  size_t i = (size_t)blockIdx.x * 256 + threadIdx.x;
  if (i >= total) return;
  const size_t row = i / 544;
  const int c = (int)(i % 544);
  float m = 0.f;
  if (c < 513) {
    const float re = (float)spec[row * 1056 + c];
    const float im = (float)spec[row * 1056 + 513 + c];
    m = sqrtf(fmaxf(re * re + im * im, 1.1920929e-7f));
  }
  mag[i] = (_Float16)m;
}

__global__ void k_maskapply(_Float16* __restrict__ spec,
                            const _Float16* __restrict__ mask, size_t total)
{
  size_t i = (size_t)blockIdx.x * 256 + threadIdx.x;
  if (i >= total) return;
  const size_t row = i / 513;
  const int c = (int)(i % 513);
  const float mk = (float)mask[row * 544 + c];
  spec[row * 1056 + c]       = (_Float16)(mk * (float)spec[row * 1056 + c]);
  spec[row * 1056 + 513 + c] = (_Float16)(mk * (float)spec[row * 1056 + 513 + c]);
}

__global__ __launch_bounds__(256) void k_iln(
    const float* __restrict__ enc, const float* __restrict__ gamma,
    const float* __restrict__ beta, _Float16* __restrict__ out, int rows)
{
  const int row = blockIdx.x * 8 + (threadIdx.x >> 5);
  const int lane = threadIdx.x & 31;
  if (row >= rows) return;
  const float* e = enc + (size_t)row * 256;
  float s = 0.f, ss = 0.f;
  for (int c = lane; c < 256; c += 32) { const float v = e[c]; s += v; ss += v * v; }
#pragma unroll
  for (int o = 16; o > 0; o >>= 1) { s += __shfl_xor(s, o, 32); ss += __shfl_xor(ss, o, 32); }
  const float mean = s * (1.f / 256.f);
  const float var  = ss * (1.f / 256.f) - mean * mean;
  const float rstd = rsqrtf(var + 1e-7f);
  for (int c = lane; c < 256; c += 32)
    out[(size_t)row * 256 + c] = (_Float16)((e[c] - mean) * rstd * gamma[c] + beta[c]);
}

__global__ void k_me(const _Float16* __restrict__ mask2, const float* __restrict__ enc,
                     _Float16* __restrict__ me, size_t total)
{
  size_t i = (size_t)blockIdx.x * 256 + threadIdx.x;
  if (i >= total) return;
  me[i] = (_Float16)((float)mask2[i] * enc[i]);
}

// overlap-add as a 4-tap gather (hop = frame/4): deterministic, no atomics
__global__ void k_ola(const _Float16* __restrict__ dec, float* __restrict__ out,
                      int T, int NS, size_t total)
{
  size_t i = (size_t)blockIdx.x * 256 + threadIdx.x;
  if (i >= total) return;
  const int n = (int)(i % NS);
  const int b = (int)(i / NS);
  const int t0 = n >> 8;
  float s = 0.f;
#pragma unroll
  for (int k = 0; k < 4; ++k) {
    const int t = t0 - k;
    if (t >= 0 && t < T) {
      const int off = n - t * 256;               // always < 1024 here
      s += (float)dec[((size_t)b * T + t) * 1024 + off];
    }
  }
  out[i] = s;
}

// forward rDFT matrix, stored Bt (N=1056 x K=1024): rows<513 cos, rows<1026 -sin
__global__ void k_gen_dtt(_Float16* __restrict__ D)
{
  const int i = blockIdx.x * 256 + threadIdx.x;
  if (i >= 1056 * 1024) return;
  const int n = i >> 10, k = i & 1023;
  float v = 0.f;
  if (n < 513)       { const int m = (n * k) & 1023;         v =  __cosf((float)m * TWO_PI_OVER_1024); }
  else if (n < 1026) { const int m = ((n - 513) * k) & 1023;  v = -__sinf((float)m * TWO_PI_OVER_1024); }
  D[i] = (_Float16)v;
}

// inverse rDFT matrix, stored Bt (N=1024 x K=1056)
__global__ void k_gen_dinvt(_Float16* __restrict__ D)
{
  const int i = blockIdx.x * 256 + threadIdx.x;
  if (i >= 1024 * 1056) return;
  const int j = i / 1056, k = i % 1056;
  float v = 0.f;
  if (k < 513) {
    const float w = (k == 0 || k == 512) ? 1.f : 2.f;
    const int m = (j * k) & 1023;
    v = w * __cosf((float)m * TWO_PI_OVER_1024) * (1.f / 1024.f);
  } else if (k < 1026) {
    const int kk = k - 513;
    const float w = (kk == 0 || kk == 512) ? 1.f : 2.f;
    const int m = (j * kk) & 1023;
    v = -w * __sinf((float)m * TWO_PI_OVER_1024) * (1.f / 1024.f);
  }
  D[i] = (_Float16)v;
}

// f32 -> f16 weight copy with row/col zero padding (dst is dstRows x ld)
__global__ void k_wconv(_Float16* __restrict__ dst, const float* __restrict__ src,
                        int srcRows, int srcCols, int ld, size_t total)
{
  size_t i = (size_t)blockIdx.x * 256 + threadIdx.x;
  if (i >= total) return;
  const int r = (int)(i / ld), c = (int)(i % ld);
  const float v = (r < srcRows && c < srcCols) ? src[(size_t)r * srcCols + c] : 0.f;
  dst[i] = (_Float16)v;
}

__global__ void k_badd(float* __restrict__ d, const float* __restrict__ a,
                       const float* __restrict__ b, int n)
{
  const int i = blockIdx.x * 256 + threadIdx.x;
  if (i < n) d[i] = a[i] + b[i];
}

__global__ void k_bpad(float* __restrict__ d, const float* __restrict__ s, int n, int ld)
{
  const int i = blockIdx.x * 256 + threadIdx.x;
  if (i < ld) d[i] = (i < n) ? s[i] : 0.f;
}

// ---------------------------------------------------------------------------
// Host side
// ---------------------------------------------------------------------------
static inline void run_gemm(const _Float16* A, int lda, const _Float16* Bt, int ldb,
                            const float* bias, void* C, int ldc,
                            int M, int N, int K, int act, int out16, hipStream_t s)
{
  dim3 g((N + 63) / 64, (M + 127) / 128);
  if (act == 0 && out16 == 1) gemm_wmma<0, 1><<<g, 256, 0, s>>>(A, lda, Bt, ldb, bias, C, ldc, M, N, K);
  else if (act == 0)          gemm_wmma<0, 0><<<g, 256, 0, s>>>(A, lda, Bt, ldb, bias, C, ldc, M, N, K);
  else if (out16 == 1)        gemm_wmma<1, 1><<<g, 256, 0, s>>>(A, lda, Bt, ldb, bias, C, ldc, M, N, K);
  else                        gemm_wmma<1, 0><<<g, 256, 0, s>>>(A, lda, Bt, ldb, bias, C, ldc, M, N, K);
}

#define GRID1D(total) ((unsigned)(((total) + 255) / 256))

extern "C" void kernel_launch(void* const* d_in, const int* in_sizes, int n_in,
                              void* d_out, int out_size, void* d_ws, size_t ws_size,
                              hipStream_t stream)
{
  (void)in_sizes; (void)n_in; (void)out_size; (void)ws_size;
  const int Bn = 64, NS = 256000, T = (NS - 1024) / 256 + 1;  // T = 997
  const int BT = Bn * T;                                      // 63808 (x16)

  const float* x      = (const float*)d_in[0];
  const float* W1ih1s = (const float*)d_in[3];
  const float* W1hh1s = (const float*)d_in[4];
  const float* b1ih1  = (const float*)d_in[5];
  const float* b1hh1  = (const float*)d_in[6];
  const float* W1ih2s = (const float*)d_in[7];
  const float* W1hh2s = (const float*)d_in[8];
  const float* b1ih2  = (const float*)d_in[9];
  const float* b1hh2  = (const float*)d_in[10];
  const float* W1ds   = (const float*)d_in[11];
  const float* b1d    = (const float*)d_in[12];
  const float* W2ih1s = (const float*)d_in[13];
  const float* W2hh1s = (const float*)d_in[14];
  const float* b2ih1  = (const float*)d_in[15];
  const float* b2hh1  = (const float*)d_in[16];
  const float* W2ih2s = (const float*)d_in[17];
  const float* W2hh2s = (const float*)d_in[18];
  const float* b2ih2  = (const float*)d_in[19];
  const float* b2hh2  = (const float*)d_in[20];
  const float* W2ds   = (const float*)d_in[21];
  const float* b2d    = (const float*)d_in[22];
  const float* encWs  = (const float*)d_in[23];
  const float* decWs  = (const float*)d_in[24];
  const float* gamma  = (const float*)d_in[25];
  const float* beta   = (const float*)d_in[26];

  // ---- workspace carve-out (~640MB), 256B aligned ----
  char* p = (char*)d_ws;
  auto alloc = [&](size_t bytes) -> char* {
    char* r = p; p += (bytes + 255) & ~(size_t)255; return r;
  };
  _Float16* framesH = (_Float16*)alloc((size_t)BT * 1024 * 2); // reused as y1
  _Float16* specH   = (_Float16*)alloc((size_t)BT * 1056 * 2); // reused as dec
  _Float16* magH    = (_Float16*)alloc((size_t)BT * 544 * 2);  // reused as mask1
  float*    xgF     = (float*)   alloc((size_t)BT * 512 * 4);
  _Float16* hs1H    = (_Float16*)alloc((size_t)BT * 128 * 2);
  _Float16* hs2H    = (_Float16*)alloc((size_t)BT * 128 * 2);
  float*    encF    = (float*)   alloc((size_t)BT * 256 * 4);
  _Float16* encNH   = (_Float16*)alloc((size_t)BT * 256 * 2);  // reused as m*enc
  _Float16* mask2H  = (_Float16*)alloc((size_t)BT * 256 * 2);
  _Float16* DTt     = (_Float16*)alloc((size_t)1056 * 1024 * 2);
  _Float16* DinvT   = (_Float16*)alloc((size_t)1024 * 1056 * 2);
  _Float16* W1ih1f  = (_Float16*)alloc((size_t)512 * 544 * 2);
  _Float16* W1hh1f  = (_Float16*)alloc((size_t)512 * 128 * 2);
  _Float16* W1ih2f  = (_Float16*)alloc((size_t)512 * 128 * 2);
  _Float16* W1hh2f  = (_Float16*)alloc((size_t)512 * 128 * 2);
  _Float16* W1df    = (_Float16*)alloc((size_t)544 * 128 * 2);
  _Float16* W2ih1f  = (_Float16*)alloc((size_t)512 * 256 * 2);
  _Float16* W2hh1f  = (_Float16*)alloc((size_t)512 * 128 * 2);
  _Float16* W2ih2f  = (_Float16*)alloc((size_t)512 * 128 * 2);
  _Float16* W2hh2f  = (_Float16*)alloc((size_t)512 * 128 * 2);
  _Float16* W2df    = (_Float16*)alloc((size_t)256 * 128 * 2);
  _Float16* encWf   = (_Float16*)alloc((size_t)256 * 1024 * 2);
  _Float16* decWf   = (_Float16*)alloc((size_t)1024 * 256 * 2);
  float* b11 = (float*)alloc(512 * 4);
  float* b12 = (float*)alloc(512 * 4);
  float* b21 = (float*)alloc(512 * 4);
  float* b22 = (float*)alloc(512 * 4);
  float* bd1 = (float*)alloc(544 * 4);
  float* bd2 = (float*)alloc(256 * 4);

  // ---- weight prep (f32 -> f16, zero padded) ----
  k_wconv<<<GRID1D(512*544), 256, 0, stream>>>(W1ih1f, W1ih1s, 512, 513, 544, 512*544);
  k_wconv<<<GRID1D(512*128), 256, 0, stream>>>(W1hh1f, W1hh1s, 512, 128, 128, 512*128);
  k_wconv<<<GRID1D(512*128), 256, 0, stream>>>(W1ih2f, W1ih2s, 512, 128, 128, 512*128);
  k_wconv<<<GRID1D(512*128), 256, 0, stream>>>(W1hh2f, W1hh2s, 512, 128, 128, 512*128);
  k_wconv<<<GRID1D(544*128), 256, 0, stream>>>(W1df,   W1ds,   513, 128, 128, 544*128);
  k_wconv<<<GRID1D(512*256), 256, 0, stream>>>(W2ih1f, W2ih1s, 512, 256, 256, 512*256);
  k_wconv<<<GRID1D(512*128), 256, 0, stream>>>(W2hh1f, W2hh1s, 512, 128, 128, 512*128);
  k_wconv<<<GRID1D(512*128), 256, 0, stream>>>(W2ih2f, W2ih2s, 512, 128, 128, 512*128);
  k_wconv<<<GRID1D(512*128), 256, 0, stream>>>(W2hh2f, W2hh2s, 512, 128, 128, 512*128);
  k_wconv<<<GRID1D(256*128), 256, 0, stream>>>(W2df,   W2ds,   256, 128, 128, 256*128);
  k_wconv<<<GRID1D(256*1024),256, 0, stream>>>(encWf,  encWs,  256, 1024,1024,256*1024);
  k_wconv<<<GRID1D(1024*256),256, 0, stream>>>(decWf,  decWs, 1024, 256, 256,1024*256);
  k_badd<<<2, 256, 0, stream>>>(b11, b1ih1, b1hh1, 512);
  k_badd<<<2, 256, 0, stream>>>(b12, b1ih2, b1hh2, 512);
  k_badd<<<2, 256, 0, stream>>>(b21, b2ih1, b2hh1, 512);
  k_badd<<<2, 256, 0, stream>>>(b22, b2ih2, b2hh2, 512);
  k_bpad<<<3, 256, 0, stream>>>(bd1, b1d, 513, 544);
  k_bpad<<<1, 256, 0, stream>>>(bd2, b2d, 256, 256);
  k_gen_dtt<<<GRID1D(1056*1024), 256, 0, stream>>>(DTt);
  k_gen_dinvt<<<GRID1D(1024*1056), 256, 0, stream>>>(DinvT);

  // ---- pipeline ----
  const size_t totF = (size_t)BT * 1024;
  k_frames<<<GRID1D(totF), 256, 0, stream>>>(x, framesH, T, NS, totF);

  // STFT as GEMM: spec = frames x DFT  (BT x 1056, r|im|pad)
  run_gemm(framesH, 1024, DTt, 1024, nullptr, specH, 1056, BT, 1056, 1024, 0, 1, stream);
  k_mag<<<GRID1D((size_t)BT * 544), 256, 0, stream>>>(specH, magH, (size_t)BT * 544);

  // sep block 1
  run_gemm(magH, 544, W1ih1f, 544, b11, xgF, 512, BT, 512, 544, 0, 0, stream);
  lstm_scan<<<Bn / 16, 256, 0, stream>>>(xgF, W1hh1f, hs1H, T);
  run_gemm(hs1H, 128, W1ih2f, 128, b12, xgF, 512, BT, 512, 128, 0, 0, stream);
  lstm_scan<<<Bn / 16, 256, 0, stream>>>(xgF, W1hh2f, hs2H, T);
  run_gemm(hs2H, 128, W1df, 128, bd1, magH /*mask1*/, 544, BT, 544, 128, 1, 1, stream);

  // mask * spec, then irfft as GEMM -> y1 (reuses frames buffer)
  k_maskapply<<<GRID1D((size_t)BT * 513), 256, 0, stream>>>(specH, magH, (size_t)BT * 513);
  run_gemm(specH, 1056, DinvT, 1056, nullptr, framesH /*y1*/, 1024, BT, 1024, 1056, 0, 1, stream);

  // encoder + instant layer norm
  run_gemm(framesH, 1024, encWf, 1024, nullptr, encF, 256, BT, 256, 1024, 0, 0, stream);
  k_iln<<<(BT + 7) / 8, 256, 0, stream>>>(encF, gamma, beta, encNH, BT);

  // sep block 2
  run_gemm(encNH, 256, W2ih1f, 256, b21, xgF, 512, BT, 512, 256, 0, 0, stream);
  lstm_scan<<<Bn / 16, 256, 0, stream>>>(xgF, W2hh1f, hs1H, T);
  run_gemm(hs1H, 128, W2ih2f, 128, b22, xgF, 512, BT, 512, 128, 0, 0, stream);
  lstm_scan<<<Bn / 16, 256, 0, stream>>>(xgF, W2hh2f, hs2H, T);
  run_gemm(hs2H, 128, W2df, 128, bd2, mask2H, 256, BT, 256, 128, 1, 1, stream);

  // decoder: dec = (mask2*enc) x dec_W^T  (dec reuses spec buffer)
  k_me<<<GRID1D((size_t)BT * 256), 256, 0, stream>>>(mask2H, encF, encNH /*me*/, (size_t)BT * 256);
  run_gemm(encNH, 256, decWf, 256, nullptr, specH /*dec*/, 1024, BT, 1024, 256, 0, 1, stream);

  // overlap-add gather -> output (B x NS f32)
  const size_t totO = (size_t)Bn * NS;
  k_ola<<<GRID1D(totO), 256, 0, stream>>>(specH, (float*)d_out, T, NS, totO);
}
